// FFTransformer_59562606461174
// MI455X (gfx1250) — matrix operand order repaired and checked by reading, not compiled, not run on hardware
//
#include <hip/hip_runtime.h>
#include <hip/hip_bf16.h>
#include <math.h>

// ---------------------------------------------------------------------------
// MI455X / gfx1250 BitNet transformer forward.
//  - All GEMMs via V_WMMA_F32_16X16X32_BF16 (ternary weights exact in bf16).
//  - wave32, 8 waves / 256-thread block, 128x128x32 tiles.
//  - Double-buffered LDS staging via GLOBAL_LOAD_ASYNC_TO_LDS_B128 (ASYNCcnt).
// ---------------------------------------------------------------------------

typedef __attribute__((ext_vector_type(16))) __bf16 v16bf;
typedef __attribute__((ext_vector_type(8)))  float  v8f;

union FragBF16 {          // one WMMA 16-wide bf16 operand = 32B = 2 x uint4
    v16bf v;
    uint4 q[2];
};

// Wave-relative LDS byte address: addrspace(3) pointers are LDS offsets; the
// flat form carries that offset in the low 32 bits (ISA 10.2 aperture rules).
static __device__ __forceinline__ unsigned lds_addr(const void* p)
{
    return (unsigned)(unsigned long long)p;
}

// Async global -> LDS copy of 16 bytes per lane (GV mode, no VGPR data path).
static __device__ __forceinline__ void async_copy_b128(unsigned lds_dst,
                                                       const void* gsrc)
{
    asm volatile("global_load_async_to_lds_b128 %0, %1, off"
                 :: "v"(lds_dst), "v"(gsrc) : "memory");
}

static __device__ __forceinline__ void wait_asynccnt0()
{
    asm volatile("s_wait_asynccnt 0x0" ::: "memory");
}

// ---------------------------------------------------------------------------
// Generic batched GEMM:  C[m,n] = alpha * sum_k A[m,k]*B(n,k)  (+ bias[n])
//   A: bf16, row-major, leading dim lda, batch offset zo*aOut + zi*aIn
//   B: bf16. BKMAJ=false -> B stored [N][K] (weight layout W[o][d]).
//            BKMAJ=true  -> B stored [K][N] (P*V attention GEMM).
//   C: f32 or bf16 (OUTBF), leading dim ldc.
// OOB staging reads are index-clamped (EXEC stays all-ones, required for the
// async issue and for WMMA); OOB results are masked at the store.
// ---------------------------------------------------------------------------
template<bool BKMAJ, bool OUTBF, bool BIAS>
__global__ __launch_bounds__(256)
void gemm_bf16_wmma(const __bf16* __restrict__ A, int lda, long long aOut, long long aIn,
                    const __bf16* __restrict__ B, int ldb, long long bOut, long long bIn,
                    void* __restrict__ C, int ldc, long long cOut, long long cIn,
                    const float* __restrict__ bias,
                    int M, int N, int K, int nInner, float alpha)
{
    __shared__ __bf16 As[2][128 * 40];  // 128 (m) x 32 (k), stride 40 (pad)
    __shared__ __bf16 Bs[2][128 * 40];  // 128 (n) x 32 (k), stride 40 (pad)

    const int zb = blockIdx.z;
    const int zo = zb / nInner;
    const int zi = zb - zo * nInner;
    const __bf16* Ab = A + (long long)zo * aOut + (long long)zi * aIn;
    const __bf16* Bb = B + (long long)zo * bOut + (long long)zi * bIn;

    const int m0 = blockIdx.y * 128;
    const int n0 = blockIdx.x * 128;

    const int tid  = threadIdx.x;
    const int wave = tid >> 5;
    const int lane = tid & 31;
    const int wm   = wave >> 1;      // 0..3 : 32 rows each
    const int wn   = wave & 1;       // 0..1 : 64 cols each
    const int half = lane >> 4;      // lane half (WMMA operand layout)
    const int lm   = lane & 15;

    // Per-thread staging chunk coordinates (512 x 16B chunks, 2 per thread).
    const int r0  = (tid + 0)   >> 2;            // rows 0..63
    const int r1  = (tid + 256) >> 2;            // rows 64..127
    const int kk0 = (tid & 3) * 8;               // k offset within tile

    // Issue the async fills of LDS buffer `buf` for k-slice k0.
    auto issue_tile = [&](int buf, int k0) {
        // A tile
        {
            int gm0 = m0 + r0; gm0 = gm0 < M ? gm0 : M - 1;
            int gm1 = m0 + r1; gm1 = gm1 < M ? gm1 : M - 1;
            async_copy_b128(lds_addr(&As[buf][r0 * 40 + kk0]),
                            Ab + (size_t)gm0 * lda + k0 + kk0);
            async_copy_b128(lds_addr(&As[buf][r1 * 40 + kk0]),
                            Ab + (size_t)gm1 * lda + k0 + kk0);
        }
        if (!BKMAJ) {
            int gn0 = n0 + r0; gn0 = gn0 < N ? gn0 : N - 1;
            int gn1 = n0 + r1; gn1 = gn1 < N ? gn1 : N - 1;
            async_copy_b128(lds_addr(&Bs[buf][r0 * 40 + kk0]),
                            Bb + (size_t)gn0 * ldb + k0 + kk0);
            async_copy_b128(lds_addr(&Bs[buf][r1 * 40 + kk0]),
                            Bb + (size_t)gn1 * ldb + k0 + kk0);
        } else {
            // B is [K][N]: coalesced global read along n, 2B transpose into
            // LDS (sync DS stores; compiler drains DScnt before the barrier).
            int kk = tid >> 3;             // 0..31
            int nc = (tid & 7) * 16;       // 0..112
            const __bf16* src = Bb + (size_t)(k0 + kk) * ldb + n0 + nc;
            #pragma unroll
            for (int e = 0; e < 16; ++e) {
                int gn = n0 + nc + e;
                __bf16 v = (gn < N) ? src[(gn < N) ? e : 0] : (__bf16)0.0f;
                Bs[buf][(nc + e) * 40 + kk] = v;
            }
        }
    };

    v8f acc[2][4] = {};

    int buf = 0;
    issue_tile(buf, 0);                     // prologue: fill buffer 0

    for (int k0 = 0; k0 < K; k0 += 32) {
        wait_asynccnt0();                   // my copies into `buf` done
        __syncthreads();                    // everyone's copies visible

        const int nk = k0 + 32;
        if (nk < K)
            issue_tile(buf ^ 1, nk);        // prefetch next slice

        // ---- load fragments per ISA 16-bit A/B lane layouts -------------
        FragBF16 a[2];
        #pragma unroll
        for (int i = 0; i < 2; ++i) {
            int r = wm * 32 + i * 16 + lm;
            a[i].q[0] = *(const uint4*)&As[buf][r * 40 + half * 8];
            a[i].q[1] = *(const uint4*)&As[buf][r * 40 + 16 + half * 8];
        }
        FragBF16 b[4];
        #pragma unroll
        for (int j = 0; j < 4; ++j) {
            int cn = wn * 64 + j * 16 + lm;
            b[j].q[0] = *(const uint4*)&Bs[buf][cn * 40 + half * 8];
            b[j].q[1] = *(const uint4*)&Bs[buf][cn * 40 + 16 + half * 8];
        }

        // ---- 8 WMMAs: 32x64 per wave per K-step -------------------------
        #pragma unroll
        for (int i = 0; i < 2; ++i)
            #pragma unroll
            for (int j = 0; j < 4; ++j)
                acc[i][j] = __builtin_amdgcn_wmma_f32_16x16x32_bf16(
                    false, a[i].v, false, b[j].v, (short)0, acc[i][j],
                    false, false);

        buf ^= 1;
        // Next-iteration barrier guarantees all waves consumed this buffer
        // (WMMA already waited on its DS loads) before it is refilled.
    }

    // ---- store: acc element r -> row = base + r + half*8, col = base+lm --
    void* Cb = (char*)C + ((long long)zo * cOut + (long long)zi * cIn) *
                          (OUTBF ? 2 : 4);
    #pragma unroll
    for (int i = 0; i < 2; ++i) {
        #pragma unroll
        for (int j = 0; j < 4; ++j) {
            int row0 = m0 + wm * 32 + i * 16 + half * 8;
            int col  = n0 + wn * 64 + j * 16 + lm;
            if (col < N) {
                float bv = BIAS ? bias[col] : 0.0f;
                #pragma unroll
                for (int r = 0; r < 8; ++r) {
                    int row = row0 + r;
                    if (row < M) {
                        float val = acc[i][j][r] * alpha + bv;
                        if (OUTBF)
                            ((__bf16*)Cb)[(size_t)row * ldc + col] = (__bf16)val;
                        else
                            ((float*)Cb)[(size_t)row * ldc + col] = val;
                    }
                }
            }
        }
    }
}

// ---------------------------------------------------------------------------
// Elementwise / reduction kernels
// ---------------------------------------------------------------------------
__global__ __launch_bounds__(256)
void ternarize_bf16(const float* __restrict__ w, __bf16* __restrict__ o, long long n)
{
    for (long long i = (long long)blockIdx.x * 256 + threadIdx.x; i < n;
         i += (long long)gridDim.x * 256) {
        float v = w[i];
        float q = (fabsf(v) > 0.05f) ? (v > 0.0f ? 1.0f : -1.0f) : 0.0f;
        o[i] = (__bf16)q;
    }
}

__global__ __launch_bounds__(256)
void embed_pe(const int* __restrict__ tok, const float* __restrict__ emb,
              float* __restrict__ x, int T, int D, long long n)
{
    for (long long i = (long long)blockIdx.x * 256 + threadIdx.x; i < n;
         i += (long long)gridDim.x * 256) {
        int m = (int)(i / D);
        int d = (int)(i - (long long)m * D);
        int t = m % T;
        float div = expf((float)(2 * (d >> 1)) * (-9.210340371976184f / (float)D));
        float ang = (float)t * div;
        float pe  = (d & 1) ? cosf(ang) : sinf(ang);
        x[i] = emb[(size_t)tok[m] * D + d] + pe;
    }
}

__global__ __launch_bounds__(256)
void layernorm_to_bf16(const float* __restrict__ x, const float* __restrict__ g,
                       const float* __restrict__ bta, __bf16* __restrict__ out, int D)
{
    __shared__ float r1[256], r2[256];
    const int row = blockIdx.x, tid = threadIdx.x;
    const float* xr = x + (size_t)row * D;
    float s = 0.f, s2 = 0.f;
    for (int d = tid; d < D; d += 256) { float v = xr[d]; s += v; s2 += v * v; }
    r1[tid] = s; r2[tid] = s2;
    __syncthreads();
    for (int w = 128; w > 0; w >>= 1) {
        if (tid < w) { r1[tid] += r1[tid + w]; r2[tid] += r2[tid + w]; }
        __syncthreads();
    }
    float mu  = r1[0] / D;
    float var = r2[0] / D - mu * mu;
    float inv = rsqrtf(var + 1e-5f);
    for (int d = tid; d < D; d += 256)
        out[(size_t)row * D + d] = (__bf16)(((xr[d] - mu) * inv) * g[d] + bta[d]);
}

__global__ __launch_bounds__(256)
void softmax_bf16_inplace(__bf16* __restrict__ p, int T)   // T == 1024
{
    __shared__ float red[256];
    __bf16* pr = p + (size_t)blockIdx.x * T;
    const int tid = threadIdx.x;
    float vals[4];
    float mx = -3.4e38f;
    #pragma unroll
    for (int i = 0; i < 4; ++i) {
        float v = (float)pr[tid + i * 256];
        vals[i] = v;
        mx = fmaxf(mx, v);
    }
    red[tid] = mx; __syncthreads();
    for (int w = 128; w > 0; w >>= 1) {
        if (tid < w) red[tid] = fmaxf(red[tid], red[tid + w]);
        __syncthreads();
    }
    mx = red[0]; __syncthreads();
    float s = 0.f;
    #pragma unroll
    for (int i = 0; i < 4; ++i) { vals[i] = expf(vals[i] - mx); s += vals[i]; }
    red[tid] = s; __syncthreads();
    for (int w = 128; w > 0; w >>= 1) {
        if (tid < w) red[tid] += red[tid + w];
        __syncthreads();
    }
    float inv = 1.0f / red[0];
    #pragma unroll
    for (int i = 0; i < 4; ++i) pr[tid + i * 256] = (__bf16)(vals[i] * inv);
}

__global__ __launch_bounds__(256)
void absmax_part(const float* __restrict__ x, long long n, float* __restrict__ part)
{
    __shared__ float red[256];
    float m = 0.f;
    for (long long i = (long long)blockIdx.x * 256 + threadIdx.x; i < n;
         i += (long long)gridDim.x * 256)
        m = fmaxf(m, fabsf(x[i]));
    red[threadIdx.x] = m; __syncthreads();
    for (int w = 128; w > 0; w >>= 1) {
        if (threadIdx.x < w)
            red[threadIdx.x] = fmaxf(red[threadIdx.x], red[threadIdx.x + w]);
        __syncthreads();
    }
    if (threadIdx.x == 0) part[blockIdx.x] = red[0];
}

__global__ __launch_bounds__(256)
void absmax_scale(const float* __restrict__ part, int nb, float* __restrict__ scale)
{
    __shared__ float red[256];
    float m = 0.f;
    for (int i = threadIdx.x; i < nb; i += 256) m = fmaxf(m, part[i]);
    red[threadIdx.x] = m; __syncthreads();
    for (int w = 128; w > 0; w >>= 1) {
        if (threadIdx.x < w)
            red[threadIdx.x] = fmaxf(red[threadIdx.x], red[threadIdx.x + w]);
        __syncthreads();
    }
    if (threadIdx.x == 0) scale[0] = fmaxf(red[0] / 7.0f, 1e-8f);
}

// fp4 STE forward value + residual add + hidden-activation output
__global__ __launch_bounds__(256)
void fp4_residual(const float* __restrict__ o, const float* __restrict__ scale,
                  float* __restrict__ x, float* __restrict__ hid, long long n)
{
    const float s = scale[0];
    for (long long i = (long long)blockIdx.x * 256 + threadIdx.x; i < n;
         i += (long long)gridDim.x * 256) {
        float q = rintf(o[i] / s);          // round-half-even
        q = fminf(7.0f, fmaxf(-7.0f, q));
        float v = q * s;
        hid[i] = v;
        x[i]  += v;
    }
}

__global__ __launch_bounds__(256)
void gelu_to_bf16(const float* __restrict__ h, __bf16* __restrict__ out, long long n)
{
    for (long long i = (long long)blockIdx.x * 256 + threadIdx.x; i < n;
         i += (long long)gridDim.x * 256) {
        float t = h[i];
        out[i] = (__bf16)(0.5f * t * (1.0f + erff(t * 0.70710678118654752f)));
    }
}

// ---------------------------------------------------------------------------
// Host orchestration
// ---------------------------------------------------------------------------
extern "C" void kernel_launch(void* const* d_in, const int* in_sizes, int n_in,
                              void* d_out, int out_size, void* d_ws, size_t ws_size,
                              hipStream_t stream)
{
    (void)in_sizes; (void)n_in; (void)out_size;
    constexpr int L = 4, D = 1024, NH = 16, DH = 64, HID = 4096, V = 32000;
    constexpr int T = 1024, Bsz = 2;
    constexpr long long MT = (long long)Bsz * T;          // 2048 rows

    const int*   tokens = (const int*)d_in[0];
    const float* emb  = (const float*)d_in[1];
    const float* ln1g = (const float*)d_in[2];
    const float* ln1b = (const float*)d_in[3];
    const float* Wq = (const float*)d_in[4];   const float* bq = (const float*)d_in[5];
    const float* Wk = (const float*)d_in[6];   const float* bk = (const float*)d_in[7];
    const float* Wv = (const float*)d_in[8];   const float* bv = (const float*)d_in[9];
    const float* Wo = (const float*)d_in[10];  const float* bo = (const float*)d_in[11];
    const float* ln2g = (const float*)d_in[12]; const float* ln2b = (const float*)d_in[13];
    const float* W1 = (const float*)d_in[14];  const float* b1 = (const float*)d_in[15];
    const float* W2 = (const float*)d_in[16];  const float* b2 = (const float*)d_in[17];
    const float* lnfg = (const float*)d_in[18]; const float* lnfb = (const float*)d_in[19];
    const float* Wr = (const float*)d_in[20];  const float* br = (const float*)d_in[21];

    // ---- workspace carve-out (aligned 256B) ----
    char*  base = (char*)d_ws;
    size_t off  = 0;
    auto alloc = [&](size_t bytes) -> char* {
        char* p = base + off;
        off += (bytes + 255) & ~(size_t)255;
        return p;
    };
    __bf16* wq_t = (__bf16*)alloc((size_t)L * D * D * 2);
    __bf16* wk_t = (__bf16*)alloc((size_t)L * D * D * 2);
    __bf16* wv_t = (__bf16*)alloc((size_t)L * D * D * 2);
    __bf16* wo_t = (__bf16*)alloc((size_t)L * D * D * 2);
    __bf16* w1_t = (__bf16*)alloc((size_t)L * HID * D * 2);
    __bf16* w2_t = (__bf16*)alloc((size_t)L * D * HID * 2);
    __bf16* wr_t = (__bf16*)alloc((size_t)V * D * 2);
    float*  x    = (float*)alloc((size_t)MT * D * 4);
    __bf16* xn   = (__bf16*)alloc((size_t)MT * D * 2);
    __bf16* qb   = (__bf16*)alloc((size_t)MT * D * 2);
    __bf16* kb   = (__bf16*)alloc((size_t)MT * D * 2);
    __bf16* vb   = (__bf16*)alloc((size_t)MT * D * 2);
    __bf16* yb   = (__bf16*)alloc((size_t)MT * D * 2);
    float*  obuf = (float*)alloc((size_t)MT * D * 4);
    float*  hbuf = (float*)alloc((size_t)MT * HID * 4);
    __bf16* hb   = (__bf16*)alloc((size_t)MT * HID * 2);
    __bf16* pbuf = (__bf16*)alloc((size_t)Bsz * NH * T * T * 2);   // scores/probs
    float*  partials = (float*)alloc(1024 * 4);
    float*  scalep   = (float*)alloc(4);
    if (off > ws_size) return;   // not enough scratch; bail safely

    float* logits = (float*)d_out;                               // [B,T,V]
    float* hidden = (float*)d_out + (size_t)Bsz * T * V;         // [2L,B,T,D]

    // ---- ternarize all weights into bf16 (exact for {-1,0,+1}) ----
    ternarize_bf16<<<4096, 256, 0, stream>>>(Wq, wq_t, (long long)L * D * D);
    ternarize_bf16<<<4096, 256, 0, stream>>>(Wk, wk_t, (long long)L * D * D);
    ternarize_bf16<<<4096, 256, 0, stream>>>(Wv, wv_t, (long long)L * D * D);
    ternarize_bf16<<<4096, 256, 0, stream>>>(Wo, wo_t, (long long)L * D * D);
    ternarize_bf16<<<4096, 256, 0, stream>>>(W1, w1_t, (long long)L * HID * D);
    ternarize_bf16<<<4096, 256, 0, stream>>>(W2, w2_t, (long long)L * D * HID);
    ternarize_bf16<<<4096, 256, 0, stream>>>(Wr, wr_t, (long long)V * D);

    // ---- embedding + sinusoidal PE ----
    embed_pe<<<4096, 256, 0, stream>>>(tokens, emb, x, T, D, MT * D);

    const dim3 gD (D / 128,  (unsigned)(MT / 128), 1);     // N=1024 GEMMs
    const dim3 gH (HID / 128,(unsigned)(MT / 128), 1);     // N=4096 GEMM
    const dim3 gS (T / 128,  T / 128, Bsz * NH);           // scores
    const dim3 gAV(1,        T / 128, Bsz * NH);           // P*V (N=64)
    const dim3 gV (V / 128,  (unsigned)(MT / 128), 1);     // LM head

    for (int i = 0; i < L; ++i) {
        // ---------------- attention ----------------
        layernorm_to_bf16<<<(unsigned)MT, 256, 0, stream>>>(x, ln1g + i * D, ln1b + i * D, xn, D);

        gemm_bf16_wmma<false, true, true><<<gD, 256, 0, stream>>>(
            xn, D, 0, 0, wq_t + (size_t)i * D * D, D, 0, 0,
            qb, D, 0, 0, bq + i * D, (int)MT, D, D, 1, 1.0f);
        gemm_bf16_wmma<false, true, true><<<gD, 256, 0, stream>>>(
            xn, D, 0, 0, wk_t + (size_t)i * D * D, D, 0, 0,
            kb, D, 0, 0, bk + i * D, (int)MT, D, D, 1, 1.0f);
        gemm_bf16_wmma<false, true, true><<<gD, 256, 0, stream>>>(
            xn, D, 0, 0, wv_t + (size_t)i * D * D, D, 0, 0,
            vb, D, 0, 0, bv + i * D, (int)MT, D, D, 1, 1.0f);

        // S[b,h,q,t] = (1/8) * q . k    (batched over z = b*NH + h)
        gemm_bf16_wmma<false, true, false><<<gS, 256, 0, stream>>>(
            qb, D, (long long)T * D, DH,
            kb, D, (long long)T * D, DH,
            pbuf, T, (long long)NH * T * T, (long long)T * T,
            nullptr, T, T, DH, NH, 0.125f);

        softmax_bf16_inplace<<<Bsz * NH * T, 256, 0, stream>>>(pbuf, T);

        // Y[b,q,h,:] = P . V   (B operand is K-major: V rows are k)
        gemm_bf16_wmma<true, true, false><<<gAV, 256, 0, stream>>>(
            pbuf, T, (long long)NH * T * T, (long long)T * T,
            vb, D, (long long)T * D, DH,
            yb, D, (long long)T * D, DH,
            nullptr, T, DH, T, NH, 1.0f);

        gemm_bf16_wmma<false, false, true><<<gD, 256, 0, stream>>>(
            yb, D, 0, 0, wo_t + (size_t)i * D * D, D, 0, 0,
            obuf, D, 0, 0, bo + i * D, (int)MT, D, D, 1, 1.0f);

        absmax_part<<<1024, 256, 0, stream>>>(obuf, MT * D, partials);
        absmax_scale<<<1, 256, 0, stream>>>(partials, 1024, scalep);
        fp4_residual<<<2048, 256, 0, stream>>>(
            obuf, scalep, x, hidden + (size_t)(2 * i) * MT * D, MT * D);

        // ---------------- FFN ----------------
        layernorm_to_bf16<<<(unsigned)MT, 256, 0, stream>>>(x, ln2g + i * D, ln2b + i * D, xn, D);

        gemm_bf16_wmma<false, false, true><<<gH, 256, 0, stream>>>(
            xn, D, 0, 0, w1_t + (size_t)i * HID * D, D, 0, 0,
            hbuf, HID, 0, 0, b1 + i * HID, (int)MT, HID, D, 1, 1.0f);

        gelu_to_bf16<<<4096, 256, 0, stream>>>(hbuf, hb, MT * HID);

        gemm_bf16_wmma<false, false, true><<<gD, 256, 0, stream>>>(
            hb, HID, 0, 0, w2_t + (size_t)i * D * HID, HID, 0, 0,
            obuf, D, 0, 0, b2 + i * D, (int)MT, D, HID, 1, 1.0f);

        absmax_part<<<1024, 256, 0, stream>>>(obuf, MT * D, partials);
        absmax_scale<<<1, 256, 0, stream>>>(partials, 1024, scalep);
        fp4_residual<<<2048, 256, 0, stream>>>(
            obuf, scalep, x, hidden + (size_t)(2 * i + 1) * MT * D, MT * D);
    }

    // ---------------- final LN + LM head ----------------
    layernorm_to_bf16<<<(unsigned)MT, 256, 0, stream>>>(x, lnfg, lnfb, xn, D);
    gemm_bf16_wmma<false, false, true><<<gV, 256, 0, stream>>>(
        xn, D, 0, 0, wr_t, D, 0, 0,
        logits, V, 0, 0, br, (int)MT, V, D, 1, 1.0f);
}
